// SymmetricNN_58153857188187
// MI455X (gfx1250) — matrix operand order repaired and checked by reading, not compiled, not run on hardware
//
#include <hip/hip_runtime.h>
#include <cstdint>

typedef __attribute__((ext_vector_type(16))) _Float16 v16h;
typedef __attribute__((ext_vector_type(2)))  _Float16 v2h;
typedef __attribute__((ext_vector_type(8)))  float    v8f;

#define H3W    1224          // W3 columns (h3 width)
#define KDIM   100           // W3 rows (h2 width)
#define NT     77            // N tiles of 16  (1232 padded)
#define KC     4             // K chunks of 32 (128 padded)
#define OUTW   2193
#define TAILW  969           // 57 * 17
#define PANEL  (NT * KC * 32 * 16)
#define TAILH  (1232 * 16)   // per-wave LDS tail buffer, halves

// Pre-swizzled f16 W3 in WMMA B-operand layout:
// element (t, kc, lane, i): K = kc*32 + (lane>=16 ? 8 : 0) + i + (i>=8 ? 8 : 0),  N = t*16 + (lane&15)
__device__ __align__(16) _Float16 g_panel[PANEL];

__global__ void pack_w3_kernel(const float* __restrict__ W3) {
    int tid = blockIdx.x * blockDim.x + threadIdx.x;
    if (tid >= PANEL) return;
    int i  =  tid        & 15;
    int L  = (tid >> 4)  & 31;
    int kc = (tid >> 9)  & 3;
    int t  =  tid >> 11;
    int K = kc * 32 + ((L >= 16) ? 8 : 0) + i + ((i >= 8) ? 8 : 0);
    int n = t * 16 + (L & 15);
    float v = (K < KDIM && n < H3W) ? W3[K * H3W + n] : 0.0f;
    g_panel[tid] = (_Float16)v;
}

// fast sigmoid: v_exp_f32 + v_rcp_f32 (1-ulp rcp, no IEEE divide expansion)
__device__ __forceinline__ float sigf(float x) {
    return __builtin_amdgcn_rcpf(1.0f + __expf(-x));
}

__global__ __launch_bounds__(128) void fused_kernel(
    const float* __restrict__ x,  const float* __restrict__ W1, const float* __restrict__ b1,
    const float* __restrict__ W2, const float* __restrict__ b2, const float* __restrict__ b3,
    const float* __restrict__ w,  const int* __restrict__ idx1, const int* __restrict__ idx2,
    float* __restrict__ out, int B)
{
    // per-wave tail buffer: h3 as f16, layout [col][row] (col 0..1231, row 0..15)
    // Total static LDS kept <= 160 KiB so two workgroups share one WGP (320 KB pool).
    __shared__ __align__(16) _Float16 s_tail[4 * TAILH];
    __shared__ float s_W2[1000], s_b2[100], s_W1[10], s_b1[10], s_w[57];
    __shared__ int   s_i1[57], s_i2[57];

    const int tid = threadIdx.x;
    for (int i = tid; i < 1000; i += 128) s_W2[i] = W2[i];
    if (tid < 100) s_b2[tid] = b2[tid];
    if (tid < 10)  { s_W1[tid] = W1[tid]; s_b1[tid] = b1[tid]; }
    if (tid < 57)  { s_w[tid] = w[tid]; s_i1[tid] = idx1[tid]; s_i2[tid] = idx2[tid]; }
    __syncthreads();

    const int wv = tid >> 5;        // wave in block
    const int L  = tid & 31;        // lane
    const int r  = L & 15;          // row within 16-row tile
    const int hi = L >> 4;          // lane half
    const int h8 = hi << 3;
    const int rowbase = blockIdx.x * 64 + wv * 16;
    const int row = rowbase + r;
    const bool full = (rowbase + 16 <= B);   // wave-uniform: all 16 rows valid

    _Float16* tb  = s_tail + wv * TAILH;   // tail buffer for this wave
    _Float16* stA = tb;                    // h2 staging (16 x 132 halves), overlaid

    // ---- layers 1 & 2 (VALU), stage h2 as f16 into LDS with padded stride ----
    {
        float xv = x[row < B ? row : (B - 1)];
        float h1[10];
        #pragma unroll
        for (int k = 0; k < 10; ++k) h1[k] = sigf(xv * s_W1[k] + s_b1[k]);
        for (int j0 = 0; j0 < 50; ++j0) {
            int j = j0 + 50 * hi;
            float acc = s_b2[j];
            #pragma unroll
            for (int k = 0; k < 10; ++k) acc += h1[k] * s_W2[k * 100 + j];
            stA[r * 132 + j] = (_Float16)sigf(acc);
        }
        if (!hi) {  // zero-pad K = 100..131
            for (int c = 100; c < 132; ++c) stA[r * 132 + c] = (_Float16)0.0f;
        }
    }
    __syncthreads();

    // ---- gather A fragments per ISA 16-bit 16x32 A layout ----
    // lane m (m=L&15): v<4 holds K = h8 + 2v, 2v+1 ; v>=4 holds K = 16 + h8 + 2(v-4), +1
    v16h a[4];
    #pragma unroll
    for (int kc = 0; kc < 4; ++kc) {
        #pragma unroll
        for (int v = 0; v < 8; ++v) {
            int K0 = kc * 32 + 2 * v + h8 + (v >= 4 ? 8 : 0);
            reinterpret_cast<uint32_t*>(&a[kc])[v] =
                *reinterpret_cast<const uint32_t*>(stA + r * 132 + K0);
        }
    }
    __syncthreads();   // everyone done reading staging before tail-buffer writes reuse it

    // ---- N-tiles: 4x WMMA (K=128), bias + sigmoid, write h3 + tail buffer ----
    const int nloc = L & 15;
    for (int t = 0; t < NT; ++t) {
        v8f c = {0.0f, 0.0f, 0.0f, 0.0f, 0.0f, 0.0f, 0.0f, 0.0f};
        #pragma unroll
        for (int kc = 0; kc < 4; ++kc) {
            v16h b = *reinterpret_cast<const v16h*>(
                g_panel + (size_t)((t * 4 + kc) * 32 + L) * 16);
            c = __builtin_amdgcn_wmma_f32_16x16x32_f16(
                    false, a[kc], false, b, (short)0, c, false, false);
        }
        if (t + 1 < NT)
            __builtin_prefetch(g_panel + (size_t)(((t + 1) * 4) * 32 + L) * 16, 0, 3);

        const int n = t * 16 + nloc;
        const float bias = b3[n < H3W ? n : (H3W - 1)];   // clamped: pad lanes read valid addr
        float vals[8];
        #pragma unroll
        for (int v = 0; v < 8; ++v) vals[v] = sigf(c[v] + bias);

        // direct f32 store of main h3 outputs (two 64B row segments per VGPR)
        if (t < NT - 1) {               // n < 1216 < H3W: no lane guard needed
            if (full) {                 // wave-uniform scalar branch, no EXEC churn
                #pragma unroll
                for (int v = 0; v < 8; ++v)
                    out[(size_t)(rowbase + v + h8) * OUTW + n] = vals[v];
            } else {
                #pragma unroll
                for (int v = 0; v < 8; ++v) {
                    int m = rowbase + v + h8;
                    if (m < B) out[(size_t)m * OUTW + n] = vals[v];
                }
            }
        } else if (n < H3W) {           // last tile: cols 1216..1223 valid
            #pragma unroll
            for (int v = 0; v < 8; ++v) {
                int m = rowbase + v + h8;
                if (m < B) out[(size_t)m * OUTW + n] = vals[v];
            }
        }
        // f16 copy into tail buffer, [col][row], packed pairs of rows
        #pragma unroll
        for (int v = 0; v < 8; v += 2) {
            v2h pk = { (_Float16)vals[v], (_Float16)vals[v + 1] };
            *reinterpret_cast<v2h*>(tb + n * 16 + v + h8) = pk;
        }
    }
    __syncthreads();

    // ---- tail: out[:, 1224 + j*17 + c] = w[j]*h3[idx1[j]*17+c] + (1-w[j])*h3[idx2[j]*17+c]
    for (int r2 = 0; r2 < 16; ++r2) {
        int orow = rowbase + r2;
        if (orow >= B) break;                    // wave-uniform
        size_t obase = (size_t)orow * OUTW + H3W;
        for (int f = L; f < TAILW; f += 32) {
            int j = f / 17;
            int cc = f - j * 17;
            float wj = s_w[j];
            float a1 = (float)tb[(s_i1[j] * 17 + cc) * 16 + r2];
            float a2 = (float)tb[(s_i2[j] * 17 + cc) * 16 + r2];
            out[obase + f] = wj * a1 + (1.0f - wj) * a2;
        }
    }
}

extern "C" void kernel_launch(void* const* d_in, const int* in_sizes, int n_in,
                              void* d_out, int out_size, void* d_ws, size_t ws_size,
                              hipStream_t stream) {
    const float* x  = (const float*)d_in[0];
    const float* W1 = (const float*)d_in[1];
    const float* b1 = (const float*)d_in[2];
    const float* W2 = (const float*)d_in[3];
    const float* b2 = (const float*)d_in[4];
    const float* W3 = (const float*)d_in[5];
    const float* b3 = (const float*)d_in[6];
    const float* w  = (const float*)d_in[7];
    const int*   i1 = (const int*)d_in[8];
    const int*   i2 = (const int*)d_in[9];
    float* out = (float*)d_out;
    int B = in_sizes[0];

    pack_w3_kernel<<<(PANEL + 255) / 256, 256, 0, stream>>>(W3);
    int nb = (B + 63) / 64;
    fused_kernel<<<nb, 128, 0, stream>>>(x, W1, b1, W2, b2, b3, w, i1, i2, out, B);
}
